// ChebConvWithSpatialAttention_28595892257030
// MI455X (gfx1250) — matrix operand order repaired and checked by reading, not compiled, not run on hardware
//
#include <hip/hip_runtime.h>
#include <hip/hip_bf16.h>

typedef __attribute__((ext_vector_type(16))) __bf16 v16bf;
typedef __attribute__((ext_vector_type(8)))  __bf16 v8bf;
typedef __attribute__((ext_vector_type(8)))  float  v8f;
typedef __attribute__((ext_vector_type(4)))  float  v4f;

#define B_    16
#define T_    12
#define N_    1024
#define FIN_  32
#define FOUT_ 64
#define K_    3
#define CDIM  (T_*FIN_)      // 384 packed columns (t*32+f)
#define CSTR  388            // padded LDS row stride (floats)
#define NTPW  3              // column tiles per wave: 24 tiles / 8 waves
#define JB    128            // j-block staged in LDS per barrier round
#define JSTEPS (JB/32)       // 4 WMMA K-steps per j-block

// ---------------------------------------------------------------------------
// Kernel 1: pack x [B,T,N,F_IN] f32  ->  xb [B, c=t*32+f, j] bf16
// Column-major-in-j so a WMMA B-operand lane load is 16 contiguous bf16.
// ---------------------------------------------------------------------------
__global__ __launch_bounds__(256) void pack_x_bf16(const float* __restrict__ x,
                                                   unsigned short* __restrict__ xb_raw) {
    __bf16* xb = reinterpret_cast<__bf16*>(xb_raw);
    int idx = blockIdx.x * 256 + threadIdx.x;        // over B*CDIM*N = 6,291,456
    int j = idx & (N_ - 1);
    int c = (idx >> 10) % CDIM;
    int b = idx / (N_ * CDIM);
    int t = c >> 5;
    int f = c & 31;
    float v = __builtin_nontemporal_load(&x[(((size_t)b * T_ + t) * N_ + j) * FIN_ + f]);
    xb[idx] = (__bf16)v;                              // xb stays L2-resident (RT)
}

// ---------------------------------------------------------------------------
// Kernel 2: fused  (cheb*att) @ X  @ Theta  + ReLU, all via v_wmma bf16.
// One workgroup (256 thr = 8 wave32) per (b, 16-row i-tile) -> 1024 WGs.
// A-operands (masked Laplacian) built cooperatively once per WG in LDS.
// ---------------------------------------------------------------------------
__global__ __launch_bounds__(256) void cheb_attn_wmma(
    const float* __restrict__ att,             // [B,N,N]
    const float* __restrict__ cheb,            // [K,N,N]
    const float* __restrict__ theta,           // [K,F_IN,F_OUT]
    const unsigned short* __restrict__ xb_raw, // [B,CDIM,N] bf16
    float* __restrict__ out) {                 // [B,T,N,F_OUT]

    const __bf16* xb = reinterpret_cast<const __bf16*>(xb_raw);

    // A-stage: [k][step][lane][16 bf16] in per-lane A-operand order, contiguous k = 16*half + s
    __shared__ alignas(32) __bf16 a_lds[K_ * JSTEPS * 32 * 16];   // 12,288 B
    __shared__ float  rhs_lds[K_ * 16 * CSTR];                    // 74,496 B
    __shared__ __bf16 thT[K_ * FOUT_ * FIN_];                     // 12,288 B [k][o][f]

    const int b     = blockIdx.x >> 6;
    const int itile = blockIdx.x & 63;
    const int i0    = itile * 16;
    const int lane  = threadIdx.x & 31;
    const int wave  = threadIdx.x >> 5;
    const int row   = lane & 15;
    const int half  = lane >> 4;

    // Stage Theta transposed as bf16: thT[(k*64+o)*32 + f]
    for (int idx = threadIdx.x; idx < K_ * FOUT_ * FIN_; idx += 256) {
        int f = idx & 31;
        int o = (idx >> 5) & 63;
        int k = idx >> 11;
        thT[idx] = (__bf16)theta[((size_t)k * FIN_ + f) * FOUT_ + o];
    }

    // Builder-role decode: thread -> (rowb, 8-wide j group)
    const int rowb = threadIdx.x >> 4;    // 0..15  (M row within tile)
    const int jg   = threadIdx.x & 15;    // 0..15  (8 j's each -> 128 j per block)
    const int jl   = jg * 8;              // local j within block
    const int s_b  = jl >> 5;             // which 32-wide WMMA step
    const int jj   = jl & 31;             // j within step
    const int laneT = rowb + 16 * (jj >> 4);         // target lane
    const int posT  = jj & 15;                       // target element base (0 or 8)

    const float*  attB  = att + ((size_t)b * N_ + i0) * N_;
    const float*  chebB = cheb + (size_t)i0 * N_;
    const __bf16* xbB   = xb + (size_t)b * CDIM * N_;

    v8f acc[K_][NTPW];
#pragma unroll
    for (int k = 0; k < K_; ++k)
#pragma unroll
        for (int n = 0; n < NTPW; ++n)
            acc[k][n] = (v8f){0.f, 0.f, 0.f, 0.f, 0.f, 0.f, 0.f, 0.f};

    // ---- GEMM 1: rhs[k, i(16), c(384)] = (cheb_k * att) @ Xpacked ----
    for (int jb = 0; jb < N_ / JB; ++jb) {
        const int j0b = jb * JB;

        // Build masked-Laplacian A tiles once per WG (att row used once -> NT)
        {
            const float* aP = attB + (size_t)rowb * N_ + j0b + jl;
            v4f av0 = __builtin_nontemporal_load((const v4f*)aP);
            v4f av1 = __builtin_nontemporal_load((const v4f*)(aP + 4));
#pragma unroll
            for (int k = 0; k < K_; ++k) {
                const float* cP = chebB + (size_t)k * N_ * N_ + (size_t)rowb * N_ + j0b + jl;
                v4f c0 = *(const v4f*)cP;
                v4f c1 = *(const v4f*)(cP + 4);
                v8bf pk;
                pk[0] = (__bf16)(av0[0] * c0[0]); pk[1] = (__bf16)(av0[1] * c0[1]);
                pk[2] = (__bf16)(av0[2] * c0[2]); pk[3] = (__bf16)(av0[3] * c0[3]);
                pk[4] = (__bf16)(av1[0] * c1[0]); pk[5] = (__bf16)(av1[1] * c1[1]);
                pk[6] = (__bf16)(av1[2] * c1[2]); pk[7] = (__bf16)(av1[3] * c1[3]);
                *(v8bf*)&a_lds[((k * JSTEPS + s_b) * 32 + laneT) * 16 + posT] = pk;
            }
        }
        __syncthreads();

        // Consume: 4 K-steps x (3 k x 3 column-tiles) WMMAs per wave
#pragma unroll
        for (int s = 0; s < JSTEPS; ++s) {
            const int j0 = j0b + s * 32;

            v16bf amat[K_];
#pragma unroll
            for (int k = 0; k < K_; ++k)
                amat[k] = *(const v16bf*)&a_lds[((k * JSTEPS + s) * 32 + lane) * 16];

            v16bf bmat[NTPW];
#pragma unroll
            for (int n = 0; n < NTPW; ++n) {
                int c = (wave * NTPW + n) * 16 + row;     // packed column
                bmat[n] = *(const v16bf*)(xbB + (size_t)c * N_ + j0 + half * 16);
            }

#pragma unroll
            for (int k = 0; k < K_; ++k)
#pragma unroll
                for (int n = 0; n < NTPW; ++n)
                    acc[k][n] = __builtin_amdgcn_wmma_f32_16x16x32_bf16(
                        false, amat[k], false, bmat[n],
                        (short)0, acc[k][n], false, false);
        }
        __syncthreads();   // a_lds safe to overwrite next block
    }

    // ---- Spill rhs C-tiles to LDS: rhs_lds[(k*16+m)*CSTR + c] ----
#pragma unroll
    for (int k = 0; k < K_; ++k)
#pragma unroll
        for (int n = 0; n < NTPW; ++n) {
            int c = (wave * NTPW + n) * 16 + row;
#pragma unroll
            for (int r = 0; r < 8; ++r) {
                int m = r + half * 8;
                rhs_lds[(k * 16 + m) * CSTR + c] = acc[k][n][r];
            }
        }
    __syncthreads();

    // ---- GEMM 2: out[b,t,i0:i0+16,o] = ReLU( sum_k rhs_k[i, t*32:+32] @ Theta_k ) ----
#pragma unroll
    for (int q = 0; q < 6; ++q) {
        int id = wave * 6 + q;        // 48 tiles: (t, o_tile)
        int t  = id >> 2;
        int ot = id & 3;

        v8f accO = (v8f){0.f, 0.f, 0.f, 0.f, 0.f, 0.f, 0.f, 0.f};
#pragma unroll
        for (int k = 0; k < K_; ++k) {
            const float* rrow = &rhs_lds[(k * 16 + row) * CSTR + t * 32];
            v4f r0 = *(const v4f*)(rrow + half * 16);
            v4f r1 = *(const v4f*)(rrow + half * 16 + 4);
            v4f r2 = *(const v4f*)(rrow + half * 16 + 8);
            v4f r3 = *(const v4f*)(rrow + half * 16 + 12);
            v16bf a2;
            a2[0]  = (__bf16)r0[0]; a2[1]  = (__bf16)r0[1];
            a2[2]  = (__bf16)r0[2]; a2[3]  = (__bf16)r0[3];
            a2[4]  = (__bf16)r1[0]; a2[5]  = (__bf16)r1[1];
            a2[6]  = (__bf16)r1[2]; a2[7]  = (__bf16)r1[3];
            a2[8]  = (__bf16)r2[0]; a2[9]  = (__bf16)r2[1];
            a2[10] = (__bf16)r2[2]; a2[11] = (__bf16)r2[3];
            a2[12] = (__bf16)r3[0]; a2[13] = (__bf16)r3[1];
            a2[14] = (__bf16)r3[2]; a2[15] = (__bf16)r3[3];

            v16bf b2 = *(const v16bf*)(&thT[(k * FOUT_ + ot * 16 + row) * FIN_ + half * 16]);

            accO = __builtin_amdgcn_wmma_f32_16x16x32_bf16(
                false, a2, false, b2, (short)0, accO, false, false);
        }

        int o = ot * 16 + row;
#pragma unroll
        for (int r = 0; r < 8; ++r) {
            int m = r + half * 8;
            float v = accO[r];
            v = v > 0.f ? v : 0.f;
            __builtin_nontemporal_store(
                v, &out[(((size_t)b * T_ + t) * N_ + i0 + m) * FOUT_ + o]);
        }
    }
}

// ---------------------------------------------------------------------------
extern "C" void kernel_launch(void* const* d_in, const int* in_sizes, int n_in,
                              void* d_out, int out_size, void* d_ws, size_t ws_size,
                              hipStream_t stream) {
    const float* x     = (const float*)d_in[0];   // [B,T,N,F_IN]
    const float* att   = (const float*)d_in[1];   // [B,N,N]
    const float* cheb  = (const float*)d_in[2];   // [K,N,N]
    const float* theta = (const float*)d_in[3];   // [K,F_IN,F_OUT]
    float* out = (float*)d_out;                   // [B,T,N,F_OUT]

    unsigned short* xb = (unsigned short*)d_ws;   // bf16 packed x: 12.6 MB

    // Kernel 1: pack/convert x
    {
        int total = B_ * CDIM * N_;
        pack_x_bf16<<<total / 256, 256, 0, stream>>>(x, xb);
    }
    // Kernel 2: fused attention-masked Chebyshev conv via WMMA
    {
        dim3 grid(B_ * (N_ / 16));                // 1024 workgroups
        cheb_attn_wmma<<<grid, 256, 0, stream>>>(att, cheb, theta, xb, out);
    }
    (void)in_sizes; (void)n_in; (void)out_size; (void)ws_size;
}
